// MonarchMixerLayer_24558622998778
// MI455X (gfx1250) — matrix operand order repaired and checked by reading, not compiled, not run on hardware
//
#include <hip/hip_runtime.h>

typedef __attribute__((ext_vector_type(16))) _Float16 v16h;
typedef __attribute__((ext_vector_type(8)))  _Float16 v8h;
typedef __attribute__((ext_vector_type(8)))  float    v8f;

#define SS        32     // monarch block size (sqrt of vector length)
#define VEC       1024   // vector length
#define NV        64     // vectors per workgroup
#define ROWP      1032   // padded LDS row length in halves (mult. of 8 -> rows stay 16B aligned)
#define NTHREADS  128    // 4 waves of 32

// involutive 32x32 transpose of a flat index in [0,1024)
__device__ __forceinline__ int perm32(int i) { return ((i & 31) << 5) | (i >> 5); }

// One block-diagonal step. The input buffer holds the step's operand in the
// PERMUTED domain (u = perm(v), stored naturally), so the A-operand read is
// fully contiguous: per CDNA5 16-bit A layout, lane L (g=L>>4) needs
// K = 8g..8g+7 (halves 0-7) and K = 16+8g..23+8g (halves 8-15) -> two aligned
// ds_load_b128 per lane.
//   TSTORE=true  (step A): store D element (a,n) at [n*32+a]  -> buffer becomes
//                          perm(w1), i.e. step B's natural input.
//   TSTORE=false (step B): store naturally at [a*32+n]        -> buffer becomes
//                          perm(monarch output) = next monarch's natural input.
// wpack holds B operands in CDNA5 16-bit B layout:
//   offset = ((a*2 + tile)*32 + lane)*16 + s, value = W[a][n = tile*16+(lane&15)][k = table order of (lane>>4, s)]
template<bool TSTORE>
__device__ __forceinline__ void blockdiag_step(
    const _Float16* __restrict__ bufIn,
    _Float16*       __restrict__ bufOut,
    const _Float16* __restrict__ wpack,
    int wavebase, int lane)
{
  const int g    = lane >> 4;
  const int mrow = lane & 15;    // A-matrix row M = vector index within the wave's 16
  for (int a = 0; a < SS; ++a) {
    const _Float16* arow = bufIn + (size_t)(wavebase + mrow) * ROWP + a * SS;
    const v8h lo = *(const v8h*)(arow + 8 * g);        // halves 0..7  : K = 8g+0..7
    const v8h hi = *(const v8h*)(arow + 16 + 8 * g);   // halves 8..15 : K = 16+8g+0..7
    const v16h amat = __builtin_shufflevector(lo, hi,
        0, 1, 2, 3, 4, 5, 6, 7, 8, 9, 10, 11, 12, 13, 14, 15);
    #pragma unroll
    for (int t = 0; t < 2; ++t) {
      // B operand: each lane loads 16 contiguous halves (32B) from packed weights
      const v16h bmat = *(const v16h*)(wpack + ((size_t)(a * 2 + t) * 32 + lane) * 16);
      v8f c = {};
      c = __builtin_amdgcn_wmma_f32_16x16x32_f16(
              /*neg_a=*/false, amat, /*neg_b=*/false, bmat,
              /*c_mod=*/(short)0, c, /*reuse_a=*/false, /*reuse_b=*/false);
      // D layout: VGPR r -> row M = r + 8*g, col N = (lane&15); tile t shifts N by 16
      const int n = mrow + 16 * t;
      #pragma unroll
      for (int r = 0; r < 8; ++r) {
        const int vrow = r + 8 * g;
        const size_t off = (size_t)(wavebase + vrow) * ROWP +
                           (TSTORE ? (n * SS + a) : (a * SS + n));
        bufOut[off] = (_Float16)c[r];
      }
    }
  }
}

// Repack the 8 (32,32,32) f32 block-diagonal weights into f16 WMMA-B layout.
__global__ __launch_bounds__(256)
void repack_weights(const float* __restrict__ w0, const float* __restrict__ w1,
                    const float* __restrict__ w2, const float* __restrict__ w3,
                    const float* __restrict__ w4, const float* __restrict__ w5,
                    const float* __restrict__ w6, const float* __restrict__ w7,
                    _Float16* __restrict__ wp)
{
  const int idx = blockIdx.x * 256 + threadIdx.x;   // 8*32768 total
  const float* ws[8] = {w0, w1, w2, w3, w4, w5, w6, w7};
  const int wi  = idx >> 15;
  const int rem = idx & 32767;
  const int a   = rem >> 10;
  const int r2  = rem & 1023;
  const int t   = r2 >> 9;
  const int k   = ((r2 & 256) >> 5) + ((r2 & 8) << 1) + (r2 & 7);
  const int h   = (r2 >> 4) & 15;
  const int n   = t * 16 + h;
  wp[idx] = (_Float16)ws[wi][a * 1024 + n * 32 + k];
}

// Phase 1: operate along N on x^T. Vectors are x[b, :, d] for 64 consecutive d.
__global__ __launch_bounds__(NTHREADS)
void monarch_phase1(const float* __restrict__ x,
                    const float* __restrict__ nker,
                    const _Float16* __restrict__ wpack,   // m1_L,m1_R,m2_L,m2_R packed
                    float* __restrict__ xtilde)
{
  __shared__ __attribute__((aligned(16))) _Float16 buf[2][NV][ROWP];
  const int b        = blockIdx.x >> 4;
  const int d0       = (blockIdx.x & 15) * NV;
  const int tid      = threadIdx.x;
  const int lane     = tid & 31;
  const int wavebase = (tid >> 5) * 16;

  // Cooperative load into the PERMUTED domain: buf0[vec][perm(n)] = x[b, n, d0+vec]
  for (int i = tid; i < NV * VEC; i += NTHREADS) {
    const int vec = i & (NV - 1);
    const int n   = i >> 6;
    buf[0][vec][perm32(n)] =
        (_Float16)x[((size_t)b << 20) + ((size_t)n << 10) + d0 + vec];
  }
  __syncthreads();

  // monarch 1 (wave-local: each wave touches only its own 16 rows)
  blockdiag_step<true >(&buf[0][0][0], &buf[1][0][0], wpack + 0 * 32768, wavebase, lane);
  blockdiag_step<false>(&buf[1][0][0], &buf[0][0][0], wpack + 1 * 32768, wavebase, lane);

  // relu(n_kernel[d, n] * v); buffer is in permuted domain, so element at
  // buffer index perm(e) corresponds to natural position e. Coalesced nker read.
  for (int i = lane; i < 16 * VEC; i += 32) {
    const int vec = wavebase + (i >> 10);
    const int e   = i & (VEC - 1);
    const int j   = perm32(e);
    const float v = (float)buf[0][vec][j] * nker[(size_t)(d0 + vec) * VEC + e];
    buf[0][vec][j] = (_Float16)(v > 0.f ? v : 0.f);
  }

  // monarch 2
  blockdiag_step<true >(&buf[0][0][0], &buf[1][0][0], wpack + 2 * 32768, wavebase, lane);
  blockdiag_step<false>(&buf[1][0][0], &buf[0][0][0], wpack + 3 * 32768, wavebase, lane);
  __syncthreads();

  // Store x_tilde[b, n, d0+vec] in f32 (coalesced over d); un-perm via index math
  for (int i = tid; i < NV * VEC; i += NTHREADS) {
    const int vec = i & (NV - 1);
    const int n   = i >> 6;
    xtilde[((size_t)b << 20) + ((size_t)n << 10) + d0 + vec] =
        (float)buf[0][vec][perm32(n)];
  }
}

// Phase 2: operate along D. Vectors are x_tilde[b, n, :] for 64 consecutive n.
// Includes relu(d_kernel*.), monarch4, residual and LayerNorm.
__global__ __launch_bounds__(NTHREADS)
void monarch_phase2(const float* __restrict__ xt,
                    const _Float16* __restrict__ wpack,   // m3_L,m3_R,m4_L,m4_R packed
                    const float* __restrict__ dker,
                    const float* __restrict__ gamma,
                    const float* __restrict__ beta,
                    float* __restrict__ out)
{
  __shared__ __attribute__((aligned(16))) _Float16 buf[2][NV][ROWP];
  const int b        = blockIdx.x >> 4;
  const int n0       = (blockIdx.x & 15) * NV;
  const int tid      = threadIdx.x;
  const int lane     = tid & 31;
  const int wavebase = (tid >> 5) * 16;

  // Cooperative load (coalesced over e) into permuted domain
  for (int i = tid; i < NV * VEC; i += NTHREADS) {
    const int vec = i >> 10;
    const int e   = i & (VEC - 1);
    buf[0][vec][perm32(e)] =
        (_Float16)xt[((size_t)b << 20) + ((size_t)(n0 + vec) << 10) + e];
  }
  __syncthreads();

  // monarch 3
  blockdiag_step<true >(&buf[0][0][0], &buf[1][0][0], wpack + 0 * 32768, wavebase, lane);
  blockdiag_step<false>(&buf[1][0][0], &buf[0][0][0], wpack + 1 * 32768, wavebase, lane);

  // relu(d_kernel[e] * v), wave-local, permuted-domain buffer
  for (int i = lane; i < 16 * VEC; i += 32) {
    const int vec = wavebase + (i >> 10);
    const int e   = i & (VEC - 1);
    const int j   = perm32(e);
    const float v = (float)buf[0][vec][j] * dker[e];
    buf[0][vec][j] = (_Float16)(v > 0.f ? v : 0.f);
  }

  // monarch 4
  blockdiag_step<true >(&buf[0][0][0], &buf[1][0][0], wpack + 2 * 32768, wavebase, lane);
  blockdiag_step<false>(&buf[1][0][0], &buf[0][0][0], wpack + 3 * 32768, wavebase, lane);
  __syncthreads();

  // Residual + LayerNorm, one thread per vector (reads all rows -> after barrier)
  if (tid < NV) {
    const int vec = tid;
    const size_t base = ((size_t)b << 20) + ((size_t)(n0 + vec) << 10);
    float s = 0.f, ss = 0.f;
    for (int e = 0; e < VEC; ++e) {
      const float h = (float)buf[0][vec][perm32(e)] + xt[base + e];
      buf[1][vec][e] = (_Float16)h;
      s  += h;
      ss += h * h;
    }
    const float mu  = s * (1.f / VEC);
    const float var = ss * (1.f / VEC) - mu * mu;
    const float rs  = rsqrtf(var + 1e-5f);
    for (int e = 0; e < VEC; ++e) {
      out[base + e] = ((float)buf[1][vec][e] - mu) * rs * gamma[e] + beta[e];
    }
  }
}

extern "C" void kernel_launch(void* const* d_in, const int* in_sizes, int n_in,
                              void* d_out, int out_size, void* d_ws, size_t ws_size,
                              hipStream_t stream) {
  const float* x     = (const float*)d_in[0];
  const float* m1L   = (const float*)d_in[1];
  const float* m1R   = (const float*)d_in[2];
  const float* m2L   = (const float*)d_in[3];
  const float* m2R   = (const float*)d_in[4];
  const float* m3L   = (const float*)d_in[5];
  const float* m3R   = (const float*)d_in[6];
  const float* m4L   = (const float*)d_in[7];
  const float* m4R   = (const float*)d_in[8];
  const float* nker  = (const float*)d_in[9];
  const float* dker  = (const float*)d_in[10];
  const float* gamma = (const float*)d_in[11];
  const float* beta  = (const float*)d_in[12];

  _Float16* wpack = (_Float16*)d_ws;
  const size_t wbytes  = (size_t)8 * 32768 * sizeof(_Float16);       // 512 KB
  const size_t xtbytes = (size_t)32 * 1024 * 1024 * sizeof(float);   // 128 MB

  // x_tilde intermediate: prefer workspace; else alias onto d_out (safe: each
  // phase-2 workgroup fully reads its rows before writing those same rows).
  float* xtilde = (ws_size >= wbytes + xtbytes)
                      ? (float*)((char*)d_ws + wbytes)
                      : (float*)d_out;

  repack_weights<<<1024, 256, 0, stream>>>(m1L, m1R, m2L, m2R, m3L, m3R, m4L, m4R, wpack);
  monarch_phase1<<<512, NTHREADS, 0, stream>>>(x, nker, wpack, xtilde);
  monarch_phase2<<<512, NTHREADS, 0, stream>>>(xtilde, wpack + 4 * 32768, dker, gamma, beta,
                                               (float*)d_out);
}
